// KANGPSLayer_62989990363551
// MI455X (gfx1250) — matrix operand order repaired and checked by reading, not compiled, not run on hardware
//
#include <hip/hip_runtime.h>
#include <math.h>

// ---------------------------------------------------------------------------
// KAN-GPS layer for gfx1250 (MI455X).  Matmuls via v_wmma_f32_16x16x32_f16;
// flash-attention K/V tiles staged through LDS with global_load_async_to_lds
// (ASYNCcnt double-buffering), shared by 8 query-tile waves per workgroup.
// N=4096 nodes, D=128, E=131072, HID=16, H=8 heads, HD=16.
// ---------------------------------------------------------------------------

typedef __attribute__((ext_vector_type(16))) _Float16 v16h;
typedef __attribute__((ext_vector_type(8)))  float    v8f;

#define NNODES 4096
#define DFEAT  128
#define NEDGES 131072
#define HIDF   16
#define NHEAD  8
#define HDIM   16
#define FQW    8      // q-tiles (waves) per flash workgroup

static __device__ __forceinline__ float siluf(float v) {
    return v / (1.0f + __expf(-v));
}

// Uniform-grid Cox–de Boor B-spline bases, degree KDEG, G intervals.
// grid t_j = (j - KDEG)*h - 1,  j = 0..G+2*KDEG.  Produces G+KDEG bases.
template <int G, int KDEG>
static __device__ __forceinline__ void bspline(float x, float h, float* out) {
    const int NI = G + 2 * KDEG;       // number of degree-0 bases
    float b[NI];
#pragma unroll
    for (int j = 0; j < NI; ++j) {
        float t0 = (float)(j - KDEG) * h - 1.0f;
        float t1 = t0 + h;
        b[j] = (x >= t0 && x < t1) ? 1.0f : 0.0f;
    }
#pragma unroll
    for (int d = 1; d <= KDEG; ++d) {
        float inv = 1.0f / ((float)d * h);
#pragma unroll
        for (int j = 0; j < NI - d; ++j) {
            float tj   = (float)(j - KDEG) * h - 1.0f;
            float tjd1 = tj + (float)(d + 1) * h;
            b[j] = (x - tj) * inv * b[j] + (tjd1 - x) * inv * b[j + 1];
        }
    }
#pragma unroll
    for (int j = 0; j < G + KDEG; ++j) out[j] = b[j];
}

// ------------------------- GCN aggregation ---------------------------------

__global__ void k_fill_deg(float* deg) {
    int i = blockIdx.x * blockDim.x + threadIdx.x;
    if (i < NNODES) deg[i] = 1.0f;   // self loop
}

__global__ void k_deg_edges(const int* __restrict__ row, float* deg) {
    int e = blockIdx.x * blockDim.x + threadIdx.x;
    if (e < NEDGES) atomicAdd(&deg[row[e]], 1.0f);
}

__global__ void k_dis(const float* __restrict__ deg, float* dis) {
    int i = blockIdx.x * blockDim.x + threadIdx.x;
    if (i < NNODES) {
        float d = deg[i];
        dis[i] = d > 0.0f ? rsqrtf(d) : 0.0f;
    }
}

// agg initialized with self-loop term: agg = dis^2 * x  (also zero-fills)
__global__ void k_self_agg(const float* __restrict__ x,
                           const float* __restrict__ dis, float* agg) {
    int t = blockIdx.x * blockDim.x + threadIdx.x;
    if (t < NNODES * DFEAT) {
        int n = t >> 7;
        float d = dis[n];
        agg[t] = d * d * x[t];
    }
}

// one wave per edge, float4 per lane (128 floats / 32 lanes)
__global__ void k_edge_agg(const int* __restrict__ row, const int* __restrict__ col,
                           const float* __restrict__ dis, const float* __restrict__ x,
                           float* agg) {
    int e    = blockIdx.x * (blockDim.x >> 5) + (threadIdx.x >> 5);
    int lane = threadIdx.x & 31;
    if (e >= NEDGES) return;
    int rs = row[e], cd = col[e];
    float w = dis[rs] * dis[cd];
    float4 xv = ((const float4*)(x + (size_t)rs * DFEAT))[lane];
    float* ag = agg + (size_t)cd * DFEAT + lane * 4;
    atomicAdd(&ag[0], w * xv.x);
    atomicAdd(&ag[1], w * xv.y);
    atomicAdd(&ag[2], w * xv.z);
    atomicAdd(&ag[3], w * xv.w);
}

// ----------------- KAN feature builder (WMMA A-layout tiles) ---------------
// A 16x32 tiles use contiguous k order here: lane<16 row=lane holds K 0..15;
// lane>=16 row=lane-16 holds K 16..31.
template <int IN, int NB, bool SILU, int G, int KDEG>
__global__ void k_features(const float* __restrict__ in, _Float16* __restrict__ At,
                           float hstep) {
    int t = blockIdx.x * blockDim.x + threadIdx.x;
    if (t >= NNODES * IN) return;
    int n = t / IN, i = t % IN;
    float v = in[(size_t)n * IN + i];
    float vals[NB + 1];
    vals[0] = SILU ? siluf(v) : v;
    bspline<G, KDEG>(v, hstep, vals + 1);

    const int K  = IN * (NB + 1);
    const int KT = K / 32;
    int mt = n >> 4, r = n & 15;
#pragma unroll
    for (int j = 0; j <= NB; ++j) {
        int f  = i * (NB + 1) + j;
        int kt = f >> 5, k = f & 31;
        int hi = k >> 4;
        int lane = r + (hi ? 16 : 0);
        int slot = k & 15;
        At[((size_t)(mt * KT + kt)) * 512 + lane * 16 + slot] = (_Float16)vals[j];
    }
}

// ------------- KAN weight packer (WMMA B-layout tiles) ---------------------
// B 32x16 tiles use contiguous k order here, matching A: lane<16 col=lane holds K 0..15; lane>=16 K 16..31.
template <int O, int I, int NB>
__global__ void k_pack_kan_w(const float* __restrict__ bw, const float* __restrict__ sw,
                             _Float16* __restrict__ Bt) {
    const int K = I * (NB + 1);
    int t = blockIdx.x * blockDim.x + threadIdx.x;
    if (t >= K * O) return;
    int k = t / O, n = t % O;
    int i = k / (NB + 1), j = k % (NB + 1);
    float v = (j == 0) ? bw[(size_t)n * I + i]
                       : sw[((size_t)n * I + i) * NB + (j - 1)];
    const int NT = O / 16;
    int kt = k >> 5, kk = k & 31;
    int lane = (n & 15) + ((kk >= 16) ? 16 : 0);
    int slot = kk & 15;
    int nt = n >> 4;
    Bt[((size_t)(kt * NT + nt)) * 512 + lane * 16 + slot] = (_Float16)v;
}

// fused Q|K|V weights: K=512 (=128 chans * 4), Nc=384 (Q:0..127,K:128..255,V:256..383)
__global__ void k_pack_qkv(const float* __restrict__ qbw, const float* __restrict__ qsw,
                           const float* __restrict__ kbw, const float* __restrict__ ksw,
                           const float* __restrict__ vbw, const float* __restrict__ vsw,
                           _Float16* __restrict__ Bt) {
    const int K = 512, O = 384;
    int t = blockIdx.x * blockDim.x + threadIdx.x;
    if (t >= K * O) return;
    int k = t / O, n = t % O;
    int i = k >> 2, j = k & 3;   // 1 base + 3 splines per channel
    int which = n >> 7, o = n & 127;
    const float* bw = (which == 0) ? qbw : (which == 1) ? kbw : vbw;
    const float* sw = (which == 0) ? qsw : (which == 1) ? ksw : vsw;
    float v = (j == 0) ? bw[(size_t)o * 128 + i]
                       : sw[((size_t)o * 128 + i) * 3 + (j - 1)];
    const int NT = O / 16;
    int kt = k >> 5, kk = k & 31;
    int lane = (n & 15) + ((kk >= 16) ? 16 : 0);
    int slot = kk & 15;
    int nt = n >> 4;
    Bt[((size_t)(kt * NT + nt)) * 512 + lane * 16 + slot] = (_Float16)v;
}

// ----------------------------- WMMA GEMM -----------------------------------
// C(M x Nc) = At(M x K, A-tiled f16) * Bt(K x Nc, B-tiled f16).  1 wave/tile.
__global__ void k_gemm(const _Float16* __restrict__ At, const _Float16* __restrict__ Bt,
                       float* __restrict__ C, int KT, int NT, int Nc) {
    int mt = blockIdx.x, nt = blockIdx.y;
    int lane = threadIdx.x;
    v8f acc = {};
    const _Float16* ap = At + ((size_t)mt * KT) * 512 + lane * 16;
    const _Float16* bp = Bt + (size_t)nt * 512 + lane * 16;
    for (int kt = 0; kt < KT; ++kt) {
        v16h a = *(const v16h*)ap;
        v16h b = *(const v16h*)bp;
        __builtin_prefetch(ap + 512, 0, 1);
        acc = __builtin_amdgcn_wmma_f32_16x16x32_f16(false, a, false, b,
                                                     (short)0, acc, false, false);
        ap += 512;
        bp += (size_t)NT * 512;
    }
    int col   = nt * 16 + (lane & 15);
    int mbase = mt * 16 + ((lane >> 4) ? 8 : 0);
#pragma unroll
    for (int r = 0; r < 8; ++r)
        C[(size_t)(mbase + r) * Nc + col] = acc[r];
}

// -------------------- QKV operand re-tiling for attention ------------------
__global__ void k_tile_q(const float* __restrict__ qkv, _Float16* __restrict__ Qa) {
    int t = blockIdx.x * blockDim.x + threadIdx.x;
    if (t >= 256 * NHEAD * 512) return;
    int blk = t >> 9, idx = t & 511;
    int qt = blk >> 3, h = blk & 7;
    int lane = idx >> 4, slot = idx & 15;
    int r  = lane & 15;
    int hi = slot >> 3;
    int lo = (slot & 7) + ((lane >= 16) ? 8 : 0);
    int k  = hi * 16 + lo;
    float v = (k < HDIM) ? qkv[(size_t)(qt * 16 + r) * 384 + h * HDIM + k] : 0.0f;
    Qa[t] = (_Float16)v;
}

__global__ void k_tile_k(const float* __restrict__ qkv, _Float16* __restrict__ Kb) {
    int t = blockIdx.x * blockDim.x + threadIdx.x;
    if (t >= 256 * NHEAD * 512) return;
    int blk = t >> 9, idx = t & 511;
    int kt16 = blk >> 3, h = blk & 7;
    int lane = idx >> 4, slot = idx & 15;
    int n  = lane & 15;
    int kk = (slot & 7) + ((lane >= 16) ? 8 : 0) + (slot >> 3) * 16;
    float v = (kk < HDIM) ? qkv[(size_t)(kt16 * 16 + n) * 384 + 128 + h * HDIM + kk] : 0.0f;
    Kb[t] = (_Float16)v;
}

__global__ void k_tile_v(const float* __restrict__ qkv, _Float16* __restrict__ Vb) {
    int t = blockIdx.x * blockDim.x + threadIdx.x;
    if (t >= 128 * NHEAD * 512) return;
    int blk = t >> 9, idx = t & 511;
    int kt32 = blk >> 3, h = blk & 7;
    int lane = idx >> 4, slot = idx & 15;
    int n  = lane & 15;
    int kk = (slot & 7) + ((lane >= 16) ? 8 : 0) + (slot >> 3) * 16;
    float v = qkv[(size_t)(kt32 * 32 + kk) * 384 + 256 + h * HDIM + n];
    Vb[t] = (_Float16)v;
}

// -------------------------- Flash attention --------------------------------
// grid (32 qgroups, 8 heads), block 256 = 8 waves, each wave one 16-query tile.
// K/V operand tiles (3 KB: Kb0|Kb1|Vb) are async-DMA'd to LDS by wave 0 and
// shared by all 8 waves; double-buffered on ASYNCcnt.  Per 32-key step:
// 2 score WMMAs + online softmax (shfl in 16-lane halves) + 1 PV WMMA.
__global__ void __launch_bounds__(256)
k_flash(const _Float16* __restrict__ Qa, const _Float16* __restrict__ Kb,
        const _Float16* __restrict__ Vb, float* __restrict__ h_attn) {
    const int lane = threadIdx.x & 31;
    const int wid  = threadIdx.x >> 5;
    const int qt   = blockIdx.x * FQW + wid;
    const int h    = blockIdx.y;

    __shared__ _Float16 KV[2][1536];   // Kb0 | Kb1 | Vb, 512 halves each
    __shared__ _Float16 P[FQW][512];

    v16h qa = *(const v16h*)(Qa + ((size_t)(qt * NHEAD + h)) * 512 + lane * 16);
    float m[8], l[8];
    v8f acc = {};
#pragma unroll
    for (int r = 0; r < 8; ++r) { m[r] = -1e30f; l[r] = 0.0f; }
    const float scale = 0.25f;   // 1/sqrt(HD=16)

    const int colbase = lane & 15;
    const int rowoff  = (lane >> 4) ? 8 : 0;

    // wave 0: async-copy one 3KB K/V tile (6 x b128 per lane) into LDS buf
    auto issue = [&](int buf, int kt) {
        uint32_t l0 = (uint32_t)(uintptr_t)(&KV[buf][0]) + lane * 16;
        uint32_t l1 = l0 + 1024;
        uint32_t l2 = l0 + 2048;
        uint64_t g0 = (uint64_t)(uintptr_t)(Kb + ((size_t)((kt * 2 + 0) * NHEAD + h)) * 512) + lane * 16;
        uint64_t g1 = (uint64_t)(uintptr_t)(Kb + ((size_t)((kt * 2 + 1) * NHEAD + h)) * 512) + lane * 16;
        uint64_t gv = (uint64_t)(uintptr_t)(Vb + ((size_t)(kt * NHEAD + h)) * 512) + lane * 16;
        asm volatile("global_load_async_to_lds_b128 %0, %1, off"            :: "v"(l0), "v"(g0) : "memory");
        asm volatile("global_load_async_to_lds_b128 %0, %1, off offset:512" :: "v"(l0), "v"(g0) : "memory");
        asm volatile("global_load_async_to_lds_b128 %0, %1, off"            :: "v"(l1), "v"(g1) : "memory");
        asm volatile("global_load_async_to_lds_b128 %0, %1, off offset:512" :: "v"(l1), "v"(g1) : "memory");
        asm volatile("global_load_async_to_lds_b128 %0, %1, off"            :: "v"(l2), "v"(gv) : "memory");
        asm volatile("global_load_async_to_lds_b128 %0, %1, off offset:512" :: "v"(l2), "v"(gv) : "memory");
    };

    if (wid == 0) issue(0, 0);

    for (int kt = 0; kt < 128; ++kt) {
        int cur = kt & 1;
        if (wid == 0) {
            if (kt + 1 < 128) {
                issue(cur ^ 1, kt + 1);
                asm volatile("s_wait_asynccnt 0x6" ::: "memory");  // batch kt retired
            } else {
                asm volatile("s_wait_asynccnt 0x0" ::: "memory");
            }
        }
        __syncthreads();   // publish KV[cur] to all waves

        v16h b0 = *(const v16h*)(&KV[cur][0]    + lane * 16);
        v16h b1 = *(const v16h*)(&KV[cur][512]  + lane * 16);
        v16h vb = *(const v16h*)(&KV[cur][1024] + lane * 16);

        v8f z = {};
        v8f s0 = __builtin_amdgcn_wmma_f32_16x16x32_f16(false, qa, false, b0,
                                                        (short)0, z, false, false);
        v8f s1 = __builtin_amdgcn_wmma_f32_16x16x32_f16(false, qa, false, b1,
                                                        (short)0, z, false, false);
#pragma unroll
        for (int r = 0; r < 8; ++r) {
            float a0 = s0[r] * scale, a1 = s1[r] * scale;
            float mx = fmaxf(a0, a1);
#pragma unroll
            for (int d = 1; d < 16; d <<= 1) mx = fmaxf(mx, __shfl_xor(mx, d, 16));
            float mn = fmaxf(m[r], mx);
            float c  = __expf(m[r] - mn);
            float e0 = __expf(a0 - mn);
            float e1 = __expf(a1 - mn);
            float sm = e0 + e1;
#pragma unroll
            for (int d = 1; d < 16; d <<= 1) sm += __shfl_xor(sm, d, 16);
            l[r] = l[r] * c + sm;
            m[r] = mn;
            acc[r] *= c;
            P[wid][(r + rowoff) * 32 + colbase]      = (_Float16)e0;
            P[wid][(r + rowoff) * 32 + 16 + colbase] = (_Float16)e1;
        }
        v16h p;
        {
            int row = lane & 15;
            int kb  = (lane >> 4) ? 8 : 0;
#pragma unroll
            for (int s = 0; s < 8; ++s) {        // same-wave LDS: in order
                p[s]     = P[wid][row * 32 + kb + s];
                p[8 + s] = P[wid][row * 32 + kb + 16 + s];
            }
        }
        acc = __builtin_amdgcn_wmma_f32_16x16x32_f16(false, p, false, vb,
                                                     (short)0, acc, false, false);
        __syncthreads();   // done with KV[cur]; refill may target it next iter
    }
#pragma unroll
    for (int r = 0; r < 8; ++r)
        h_attn[(size_t)(qt * 16 + rowoff + r) * DFEAT + h * HDIM + colbase] = acc[r] / l[r];
}

// ------------------------- residual + LayerNorm ----------------------------
__global__ void k_add_ln(const float* __restrict__ a, const float* __restrict__ b,
                         const float* __restrict__ s, const float* __restrict__ bi,
                         float* __restrict__ out) {
    int row  = blockIdx.x * (blockDim.x >> 5) + (threadIdx.x >> 5);
    int lane = threadIdx.x & 31;
    if (row >= NNODES) return;
    float4 av = ((const float4*)(a + (size_t)row * DFEAT))[lane];
    float4 bv = ((const float4*)(b + (size_t)row * DFEAT))[lane];
    float v0 = av.x + bv.x, v1 = av.y + bv.y, v2 = av.z + bv.z, v3 = av.w + bv.w;
    float sum = v0 + v1 + v2 + v3;
    float sq  = v0 * v0 + v1 * v1 + v2 * v2 + v3 * v3;
#pragma unroll
    for (int d = 1; d < 32; d <<= 1) {
        sum += __shfl_xor(sum, d, 32);
        sq  += __shfl_xor(sq, d, 32);
    }
    float mean = sum * (1.0f / 128.0f);
    float var  = sq * (1.0f / 128.0f) - mean * mean;
    float inv  = rsqrtf(var + 1e-5f);
    float4 sv  = ((const float4*)s)[lane];
    float4 biv = ((const float4*)bi)[lane];
    float4 ov;
    ov.x = (v0 - mean) * inv * sv.x + biv.x;
    ov.y = (v1 - mean) * inv * sv.y + biv.y;
    ov.z = (v2 - mean) * inv * sv.z + biv.z;
    ov.w = (v3 - mean) * inv * sv.w + biv.w;
    ((float4*)(out + (size_t)row * DFEAT))[lane] = ov;
}

__global__ void k_add(const float* __restrict__ a, const float* __restrict__ b,
                      float* __restrict__ out) {
    int t = blockIdx.x * blockDim.x + threadIdx.x;
    if (t < NNODES * DFEAT) out[t] = a[t] + b[t];
}

// ---------------------------------------------------------------------------

extern "C" void kernel_launch(void* const* d_in, const int* in_sizes, int n_in,
                              void* d_out, int out_size, void* d_ws, size_t ws_size,
                              hipStream_t stream) {
    const float* x    = (const float*)d_in[0];
    const int*   ei   = (const int*)d_in[1];      // (2,E) row-major
    const float* gbw1 = (const float*)d_in[2];
    const float* gsw1 = (const float*)d_in[3];
    const float* gbw2 = (const float*)d_in[4];
    const float* gsw2 = (const float*)d_in[5];
    const float* qbw  = (const float*)d_in[6];
    const float* qsw  = (const float*)d_in[7];
    const float* kbw  = (const float*)d_in[8];
    const float* ksw  = (const float*)d_in[9];
    const float* vbw  = (const float*)d_in[10];
    const float* vsw  = (const float*)d_in[11];
    const float* fbw1 = (const float*)d_in[12];
    const float* fsw1 = (const float*)d_in[13];
    const float* fbw2 = (const float*)d_in[14];
    const float* fsw2 = (const float*)d_in[15];
    const float* ln1s = (const float*)d_in[16];
    const float* ln1b = (const float*)d_in[17];
    const float* ln2s = (const float*)d_in[18];
    const float* ln2b = (const float*)d_in[19];
    const float* ln3s = (const float*)d_in[20];
    const float* ln3b = (const float*)d_in[21];
    float* out = (float*)d_out;

    const int* row = ei;
    const int* col = ei + NEDGES;

    // ---- workspace carving (256B aligned) ----
    char* ws = (char*)d_ws;
    size_t off = 0;
    auto carve = [&](size_t bytes) -> char* {
        char* p = ws + off;
        off = (off + bytes + 255) & ~(size_t)255;
        return p;
    };
    float*    deg    = (float*)carve(NNODES * 4);
    float*    dis    = (float*)carve(NNODES * 4);
    float*    agg    = (float*)carve((size_t)NNODES * DFEAT * 4);
    _Float16* featK  = (_Float16*)carve((size_t)NNODES * 1024 * 2); // GCN/FFN layer1 feats
    float*    h1     = (float*)carve((size_t)NNODES * HIDF * 4);
    _Float16* feat2  = (_Float16*)carve((size_t)NNODES * 128 * 2);
    float*    tmpD   = (float*)carve((size_t)NNODES * DFEAT * 4);   // gcn_out / ffn_out
    float*    hlocal = (float*)carve((size_t)NNODES * DFEAT * 4);
    _Float16* featA  = (_Float16*)carve((size_t)NNODES * 512 * 2);
    float*    qkv    = (float*)carve((size_t)NNODES * 384 * 4);
    _Float16* Qa     = (_Float16*)carve((size_t)256 * NHEAD * 512 * 2);
    _Float16* Kb     = (_Float16*)carve((size_t)256 * NHEAD * 512 * 2);
    _Float16* Vb     = (_Float16*)carve((size_t)128 * NHEAD * 512 * 2);
    float*    hattn  = (float*)carve((size_t)NNODES * DFEAT * 4);
    float*    hbuf   = (float*)carve((size_t)NNODES * DFEAT * 4);
    _Float16* W1p    = (_Float16*)carve((size_t)1024 * 16 * 2);
    _Float16* W2p    = (_Float16*)carve((size_t)128 * 128 * 2);
    _Float16* Wqkv   = (_Float16*)carve((size_t)512 * 384 * 2);
    _Float16* Wf1p   = (_Float16*)carve((size_t)1024 * 16 * 2);
    _Float16* Wf2p   = (_Float16*)carve((size_t)128 * 128 * 2);
    if (off > ws_size) return;   // insufficient scratch

    const int B = 256;
    auto cdiv = [](int a, int b) { return (a + b - 1) / b; };

    // ---- GCN aggregation ----
    k_fill_deg<<<cdiv(NNODES, B), B, 0, stream>>>(deg);
    k_deg_edges<<<cdiv(NEDGES, B), B, 0, stream>>>(row, deg);
    k_dis<<<cdiv(NNODES, B), B, 0, stream>>>(deg, dis);
    k_self_agg<<<cdiv(NNODES * DFEAT, B), B, 0, stream>>>(x, dis, agg);
    k_edge_agg<<<cdiv(NEDGES, 8), B, 0, stream>>>(row, col, dis, x, agg);

    // ---- GCN ekan2: layer1 (K=1024 -> 16), layer2 (K=128 -> 128) ----
    k_features<128, 7, true, 4, 3><<<cdiv(NNODES * 128, B), B, 0, stream>>>(agg, featK, 0.5f);
    k_pack_kan_w<16, 128, 7><<<cdiv(1024 * 16, B), B, 0, stream>>>(gbw1, gsw1, W1p);
    k_gemm<<<dim3(NNODES / 16, 1), 32, 0, stream>>>(featK, W1p, h1, 32, 1, 16);
    k_features<16, 7, true, 4, 3><<<cdiv(NNODES * 16, B), B, 0, stream>>>(h1, feat2, 0.5f);
    k_pack_kan_w<128, 16, 7><<<cdiv(128 * 128, B), B, 0, stream>>>(gbw2, gsw2, W2p);
    k_gemm<<<dim3(NNODES / 16, 8), 32, 0, stream>>>(feat2, W2p, tmpD, 4, 8, 128);
    k_add_ln<<<cdiv(NNODES, 8), B, 0, stream>>>(x, tmpD, ln1s, ln1b, hlocal);

    // ---- fused QKV KAN linear (no silu, attention grid g=2,k=1) ----
    k_features<128, 3, false, 2, 1><<<cdiv(NNODES * 128, B), B, 0, stream>>>(x, featA, 1.0f);
    k_pack_qkv<<<cdiv(512 * 384, B), B, 0, stream>>>(qbw, qsw, kbw, ksw, vbw, vsw, Wqkv);
    k_gemm<<<dim3(NNODES / 16, 24), 32, 0, stream>>>(featA, Wqkv, qkv, 16, 24, 384);

    // ---- retile QKV into WMMA operands ----
    k_tile_q<<<cdiv(256 * NHEAD * 512, B), B, 0, stream>>>(qkv, Qa);
    k_tile_k<<<cdiv(256 * NHEAD * 512, B), B, 0, stream>>>(qkv, Kb);
    k_tile_v<<<cdiv(128 * NHEAD * 512, B), B, 0, stream>>>(qkv, Vb);

    // ---- flash attention (8 q-tiles per block share async K/V staging) ----
    k_flash<<<dim3(NNODES / 16 / FQW, NHEAD), 256, 0, stream>>>(Qa, Kb, Vb, hattn);
    k_add_ln<<<cdiv(NNODES, 8), B, 0, stream>>>(x, hattn, ln2s, ln2b, hattn);

    // ---- combine branches ----
    k_add<<<cdiv(NNODES * DFEAT, B), B, 0, stream>>>(hlocal, hattn, hbuf);

    // ---- FFN ekan2 + final LN ----
    k_features<128, 7, true, 4, 3><<<cdiv(NNODES * 128, B), B, 0, stream>>>(hbuf, featK, 0.5f);
    k_pack_kan_w<16, 128, 7><<<cdiv(1024 * 16, B), B, 0, stream>>>(fbw1, fsw1, Wf1p);
    k_gemm<<<dim3(NNODES / 16, 1), 32, 0, stream>>>(featK, Wf1p, h1, 32, 1, 16);
    k_features<16, 7, true, 4, 3><<<cdiv(NNODES * 16, B), B, 0, stream>>>(h1, feat2, 0.5f);
    k_pack_kan_w<128, 16, 7><<<cdiv(128 * 128, B), B, 0, stream>>>(fbw2, fsw2, Wf2p);
    k_gemm<<<dim3(NNODES / 16, 8), 32, 0, stream>>>(feat2, Wf2p, tmpD, 4, 8, 128);
    k_add_ln<<<cdiv(NNODES, 8), B, 0, stream>>>(hbuf, tmpD, ln3s, ln3b, out);
}
